// BidirRWKV6ConvTimeMix_80436147520032
// MI455X (gfx1250) — matrix operand order repaired and checked by reading, not compiled, not run on hardware
//
#include <hip/hip_runtime.h>

// ---------------------------------------------------------------- constants
constexpr int CB  = 8;
constexpr int CT  = 2048;
constexpr int CD  = 1024;
constexpr int CH  = 16;
constexpr int CK  = 64;
constexpr int CBT = CB * CT;          // 16384 tokens
constexpr int CL  = 160;              // LoRA width 5*32
constexpr int CLP = 192;              // CL padded to 64

// activation codes
#define ACT_NONE    0
#define ACT_TANH    1
#define ACT_SILU    2
#define ACT_SIGMOID 3
#define ACT_NEGEXP  4

// ---------------------------------------------------------------- types
typedef __attribute__((ext_vector_type(16))) __bf16 v16bf;
typedef __attribute__((ext_vector_type(8)))  float  v8f;

union FragBF {
    v16bf v;
    uint4 q[2];
    unsigned short s[16];
};

__device__ __forceinline__ unsigned short f2bf(float f) {
    unsigned int u = __float_as_uint(f);
    u += 0x7FFFu + ((u >> 16) & 1u);          // round-to-nearest-even
    return (unsigned short)(u >> 16);
}

__device__ __forceinline__ float apply_act(float v, int act) {
    switch (act) {
        case ACT_TANH:    return tanhf(v);
        case ACT_SILU:    return v / (1.f + __expf(-v));
        case ACT_SIGMOID: return 1.f / (1.f + __expf(-v));
        case ACT_NEGEXP:  return -__expf(v);
        default:          return v;
    }
}

// ------------------------------------------------- weight convert+transpose
// W[K][N] f32  ->  WT[Npad][K] bf16, zero-filled pad rows (Npad >= N).
__global__ __launch_bounds__(256) void wconv_kernel(
    const float* __restrict__ W, unsigned short* __restrict__ WT,
    int K, int N, int Npad)
{
    const size_t n = (size_t)K * Npad;
    for (size_t i = (size_t)blockIdx.x * blockDim.x + threadIdx.x; i < n;
         i += (size_t)gridDim.x * blockDim.x) {
        int k  = (int)(i / Npad);
        int nn = (int)(i % Npad);
        float v = (nn < N) ? W[(size_t)k * N + nn] : 0.f;
        WT[(size_t)nn * K + k] = f2bf(v);
    }
}

// ---------------------------------------------------------------- prep:
// dx = depthwise conv3(x) - x ; xxx = x + dx*maa_x ; emit dx(f32), dx(bf16), xxx(bf16)
__global__ __launch_bounds__(256) void prep_kernel(
    const float* __restrict__ x, const float* __restrict__ conv_w,
    const float* __restrict__ maa_x,
    float* __restrict__ dxf, unsigned short* __restrict__ dxb,
    unsigned short* __restrict__ xxxb)
{
    const size_t n = (size_t)CBT * CD;
    for (size_t i = (size_t)blockIdx.x * blockDim.x + threadIdx.x; i < n;
         i += (size_t)gridDim.x * blockDim.x) {
        int    d  = (int)(i % CD);
        size_t bt = i / CD;
        int    t  = (int)(bt % CT);
        float xc = x[i];
        float xm = (t > 0)      ? x[i - CD] : 0.f;
        float xp = (t < CT - 1) ? x[i + CD] : 0.f;
        float w0 = conv_w[d * 3 + 0], w1 = conv_w[d * 3 + 1], w2 = conv_w[d * 3 + 2];
        float dx = w0 * xm + w1 * xc + w2 * xp - xc;
        dxf[i]  = dx;
        dxb[i]  = f2bf(dx);
        xxxb[i] = f2bf(xc + dx * maa_x[d]);
    }
}

// ---------------------------------------------------------------- WMMA GEMM
// C[M,N] = act( A[M,K](bf16) @ WT[N,K](bf16, pre-transposed) + bias[N] )
// 256 threads = 8 waves (4x2), 64x64 block tile, KB=32, double-buffered LDS
// staged with CDNA5 async global->LDS DMA (ASYNCcnt / s_wait_asynccnt).
__global__ __launch_bounds__(256) void gemm_bf16_kernel(
    const unsigned short* __restrict__ A, const unsigned short* __restrict__ WT,
    float* __restrict__ Cf, unsigned short* __restrict__ Cbf,
    int M, int N, int Kdim,
    const float* __restrict__ bias, int act, int store_bf16)
{
    __shared__ unsigned short As[2][64][32];   // [buf][m][k]  2 x 4 KB
    __shared__ unsigned short Bs[2][64][32];   // [buf][n][k]  2 x 4 KB

    const int tid  = threadIdx.x;
    const int lane = tid & 31, wid = tid >> 5;
    const int half = lane >> 4, lm = lane & 15;
    const int wm = wid & 3, wn = wid >> 2;           // 4 row-waves x 2 col-waves
    const int rowBlk = blockIdx.y * 64;
    const int colBlk = blockIdx.x * 64;

    // staging coords: 256 threads x 16 B; 4 chunks of 8 bf16 per 64 B row
    const int sr = tid >> 2;                          // 0..63 (A: m, B: n)
    const int sc = (tid & 3) << 3;                    // k offset 0,8,16,24

    auto stage_async = [&](int kb, int buf) {
        const unsigned short* ga = A  + (size_t)(rowBlk + sr) * Kdim + kb * 32 + sc;
        const unsigned short* gb = WT + (size_t)(colBlk + sr) * Kdim + kb * 32 + sc;
        unsigned la = (unsigned)(unsigned long long)&As[buf][sr][sc];
        unsigned lb = (unsigned)(unsigned long long)&Bs[buf][sr][sc];
        asm volatile("global_load_async_to_lds_b128 %0, %1, off"
                     :: "v"(la), "v"((unsigned long long)ga) : "memory");
        asm volatile("global_load_async_to_lds_b128 %0, %1, off"
                     :: "v"(lb), "v"((unsigned long long)gb) : "memory");
    };

    v8f acc0, acc1;
#pragma unroll
    for (int i = 0; i < 8; ++i) { acc0[i] = 0.f; acc1[i] = 0.f; }

    const int nkb = Kdim >> 5;
    stage_async(0, 0);

    for (int kb = 0; kb < nkb; ++kb) {
        const int buf = kb & 1;
        if (kb + 1 < nkb) {
            stage_async(kb + 1, buf ^ 1);
            asm volatile("s_wait_asynccnt 0x2" ::: "memory");   // oldest 2 (this buf) done
        } else {
            asm volatile("s_wait_asynccnt 0x0" ::: "memory");
        }
        __syncthreads();

        // A fragment: lane<16 holds K {0..7,16..23}, lane>=16 holds K {8..15,24..31}
        FragBF a;
        const unsigned short* arow = &As[buf][wm * 16 + lm][0];
        a.q[0] = *(const uint4*)(arow + half * 8);
        a.q[1] = *(const uint4*)(arow + 16 + half * 8);

#pragma unroll
        for (int nt = 0; nt < 2; ++nt) {
            // B fragment: lane holds column n = lane&15, K {0..7,16..23} / {8..15,24..31} (same as A)
            FragBF b;
            const unsigned short* bcol = &Bs[buf][wn * 32 + nt * 16 + lm][0];
            b.q[0] = *(const uint4*)(bcol + half * 8);
            b.q[1] = *(const uint4*)(bcol + 16 + half * 8);
            v8f& accr = nt ? acc1 : acc0;
            accr = __builtin_amdgcn_wmma_f32_16x16x32_bf16(
                false, a.v, false, b.v, (short)0, accr, false, false);
        }
        __syncthreads();   // all waves done with `buf` before it is re-DMA'd
    }

    // epilogue: D tile layout -> M = e + half*8, N = lane&15
#pragma unroll
    for (int nt = 0; nt < 2; ++nt) {
        int col = colBlk + wn * 32 + nt * 16 + lm;
        if (col >= N) continue;
        float bb = bias ? bias[col] : 0.f;
        v8f accr = nt ? acc1 : acc0;
#pragma unroll
        for (int e = 0; e < 8; ++e) {
            int row = rowBlk + wm * 16 + e + half * 8;
            float v = apply_act(accr[e] + bb, act);
            size_t idx = (size_t)row * N + col;
            if (store_bf16) Cbf[idx] = f2bf(v);
            else            Cf[idx]  = v;
        }
    }
}

// ---------------------------------------------------------------- mix:
// m_j = t5[:, j*32:(j+1)*32] @ maa_w2[j]   (single K=32 WMMA per 16x16 tile)
// X_j = x + dx * (maa_j + m_j)  -> bf16
__global__ __launch_bounds__(32) void mix_kernel(
    const float* __restrict__ t5, const float* __restrict__ w2,
    const float* __restrict__ x, const float* __restrict__ dxf,
    const float* mw, const float* mk, const float* mv, const float* mr, const float* mg,
    unsigned short* Xw, unsigned short* Xk, unsigned short* Xv,
    unsigned short* Xr, unsigned short* Xg)
{
    const int lane = threadIdx.x & 31;
    const int half = lane >> 4, lm = lane & 15;
    const int j    = blockIdx.z;
    const int dBlk = blockIdx.x * 16;
    const int tBlk = blockIdx.y * 16;

    const float*    maa = (j == 0) ? mw : (j == 1) ? mk : (j == 2) ? mv : (j == 3) ? mr : mg;
    unsigned short* out = (j == 0) ? Xw : (j == 1) ? Xk : (j == 2) ? Xv : (j == 3) ? Xr : Xg;

    FragBF a, b;
#pragma unroll
    for (int e = 0; e < 16; ++e) {
        // A (16x32 bf16 layout): element e -> K index per ISA table
        int vg = e >> 1;
        int kk = ((vg & 4) ? 16 : 0) + (half ? 8 : 0) + ((vg & 3) << 1) + (e & 1);
        a.s[e] = f2bf(t5[(size_t)(tBlk + lm) * CL + j * 32 + kk]);
        // B (32x16 bf16 layout): K order same as A, N = lane&15
        int kb = ((e & 8) ? 16 : 0) + (half ? 8 : 0) + (e & 7);
        b.s[e] = f2bf(w2[((size_t)j * 32 + kb) * CD + dBlk + lm]);
    }
    v8f acc;
#pragma unroll
    for (int i = 0; i < 8; ++i) acc[i] = 0.f;
    acc = __builtin_amdgcn_wmma_f32_16x16x32_bf16(
        false, a.v, false, b.v, (short)0, acc, false, false);

#pragma unroll
    for (int e = 0; e < 8; ++e) {
        int tok = tBlk + e + half * 8;
        int d   = dBlk + lm;
        size_t idx = (size_t)tok * CD + d;
        out[idx] = f2bf(x[idx] + dxf[idx] * (maa[d] + acc[e]));
    }
}

// ---------------------------------------------------------------- scan:
// per (b,h): thread j owns S[:,j] (64 f32 in VGPRs).
// fwd:  S = exp(w) ⊙_row S + k v^T ; y = r @ S        (writes y)
// bwd:  y = r @ S ; S = exp(w) ⊙_row (S + k v^T)      (adds into y)
__global__ __launch_bounds__(64) void rwkv_scan_kernel(
    const float* __restrict__ r, const float* __restrict__ k,
    const float* __restrict__ v, const float* __restrict__ wl,
    float* __restrict__ y, int reverse)
{
    const int bh = blockIdx.x;
    const int b  = bh / CH, h = bh % CH;
    const int j  = threadIdx.x;                        // v index 0..63

    float S[CK];
#pragma unroll
    for (int i = 0; i < CK; ++i) S[i] = 0.f;

    __shared__ float rs[CK], ks[CK], vs[CK], ws[CK];
    const size_t base = (size_t)b * CT * CD + (size_t)h * CK;

    for (int step = 0; step < CT; ++step) {
        int t = reverse ? (CT - 1 - step) : step;
        size_t off = base + (size_t)t * CD;
        __syncthreads();
        rs[j] = r[off + j];
        ks[j] = k[off + j];
        vs[j] = v[off + j];
        ws[j] = __expf(wl[off + j]);
        __syncthreads();

        float vj  = vs[j];
        float acc = 0.f;
        if (!reverse) {
#pragma unroll
            for (int i = 0; i < CK; ++i) {
                S[i] = ws[i] * S[i] + ks[i] * vj;
                acc += rs[i] * S[i];
            }
            y[off + j] = acc;
        } else {
#pragma unroll
            for (int i = 0; i < CK; ++i) {
                acc += rs[i] * S[i];
                S[i] = ws[i] * (S[i] + ks[i] * vj);
            }
            y[off + j] += acc;
        }
    }
}

// ---------------------------------------------------------------- gate + GroupNorm + *g
// one wave32 per (token, head) group of 64; lane handles elems lane, lane+32.
__global__ __launch_bounds__(256) void gn_gate_kernel(
    const float* __restrict__ y, const float* __restrict__ gate,
    const float* __restrict__ g, const float* __restrict__ lnw,
    const float* __restrict__ lnb, unsigned short* __restrict__ z)
{
    const int lane = threadIdx.x & 31;
    const int wib  = threadIdx.x >> 5;
    size_t grp = (size_t)blockIdx.x * 8 + wib;         // token*CH + h
    size_t tok = grp >> 4;
    int    h   = (int)(grp & 15);
    size_t off = tok * CD + (size_t)h * CK;

    float a = y[off + lane]      * gate[off + lane];
    float b = y[off + lane + 32] * gate[off + lane + 32];
    float s = a + b, s2 = a * a + b * b;
#pragma unroll
    for (int m = 16; m > 0; m >>= 1) {
        s  += __shfl_xor(s,  m, 32);
        s2 += __shfl_xor(s2, m, 32);
    }
    float mu  = s * (1.f / 64.f);
    float var = s2 * (1.f / 64.f) - mu * mu;
    float inv = rsqrtf(var + 6.4e-4f);                 // eps = 1e-5 * 8^2
    int d0 = h * CK + lane, d1 = d0 + 32;
    float z0 = (a - mu) * inv * lnw[d0] + lnb[d0];
    float z1 = (b - mu) * inv * lnw[d1] + lnb[d1];
    z[off + lane]      = f2bf(z0 * g[off + lane]);
    z[off + lane + 32] = f2bf(z1 * g[off + lane + 32]);
}

// ---------------------------------------------------------------- launch
extern "C" void kernel_launch(void* const* d_in, const int* in_sizes, int n_in,
                              void* d_out, int out_size, void* d_ws, size_t ws_size,
                              hipStream_t stream)
{
    const float* x          = (const float*)d_in[0];
    const float* conv_w     = (const float*)d_in[1];
    const float* maa_x      = (const float*)d_in[2];
    const float* maa_w      = (const float*)d_in[3];
    const float* maa_k      = (const float*)d_in[4];
    const float* maa_v      = (const float*)d_in[5];
    const float* maa_r      = (const float*)d_in[6];
    const float* maa_g      = (const float*)d_in[7];
    const float* maa_w1     = (const float*)d_in[8];
    const float* maa_w2     = (const float*)d_in[9];
    const float* time_decay = (const float*)d_in[10];
    const float* decay_w1   = (const float*)d_in[11];
    const float* decay_w2   = (const float*)d_in[12];
    const float* Wr         = (const float*)d_in[13];
    const float* Wk         = (const float*)d_in[14];
    const float* Wv         = (const float*)d_in[15];
    const float* Wg         = (const float*)d_in[16];
    const float* Wo         = (const float*)d_in[17];
    const float* ln_w       = (const float*)d_in[18];
    const float* ln_b       = (const float*)d_in[19];
    const float* gate_w     = (const float*)d_in[20];
    const float* gate_b     = (const float*)d_in[21];

    // ---- workspace carve ----
    char* p = (char*)d_ws;
    auto grab = [&](size_t bytes) -> char* {
        char* r = p;
        p += (bytes + 255) & ~(size_t)255;
        return r;
    };
    const size_t EF = (size_t)CBT * CD;                // elems per activation
    float*          dxf  = (float*)grab(EF * 4);
    unsigned short* dxb  = (unsigned short*)grab(EF * 2);
    unsigned short* xxxb = (unsigned short*)grab(EF * 2);
    float*          t5   = (float*)grab((size_t)CBT * CL * 4);
    unsigned short* Xw   = (unsigned short*)grab(EF * 2);
    unsigned short* Xk   = (unsigned short*)grab(EF * 2);
    unsigned short* Xv   = (unsigned short*)grab(EF * 2);
    unsigned short* Xr   = (unsigned short*)grab(EF * 2);
    unsigned short* Xg   = (unsigned short*)grab(EF * 2);
    float*          rb   = (float*)grab(EF * 4);
    float*          kb   = (float*)grab(EF * 4);
    float*          vb   = (float*)grab(EF * 4);
    float*          wlb  = (float*)grab(EF * 4);
    float*          gb   = (float*)grab(EF * 4);
    unsigned short* ub   = (unsigned short*)grab((size_t)CBT * 64 * 2);
    float*          gateb= (float*)grab(EF * 4);
    float*          yb   = (float*)grab(EF * 4);
    unsigned short* zb   = (unsigned short*)grab(EF * 2);
    // transposed bf16 weights WT[Npad][K]
    unsigned short* WTr  = (unsigned short*)grab((size_t)CD * CD * 2);
    unsigned short* WTk  = (unsigned short*)grab((size_t)CD * CD * 2);
    unsigned short* WTv  = (unsigned short*)grab((size_t)CD * CD * 2);
    unsigned short* WTg  = (unsigned short*)grab((size_t)CD * CD * 2);
    unsigned short* WTo  = (unsigned short*)grab((size_t)CD * CD * 2);
    unsigned short* WTgate = (unsigned short*)grab((size_t)CD * CD * 2);
    unsigned short* WTm1 = (unsigned short*)grab((size_t)CLP * CD * 2);
    unsigned short* WTd1 = (unsigned short*)grab((size_t)64 * CD * 2);
    unsigned short* WTd2 = (unsigned short*)grab((size_t)CD * 64 * 2);

    // 0) weight convert + transpose (bf16, L2-resident afterwards)
    wconv_kernel<<<2048, 256, 0, stream>>>(Wr,       WTr,    CD, CD, CD);
    wconv_kernel<<<2048, 256, 0, stream>>>(Wk,       WTk,    CD, CD, CD);
    wconv_kernel<<<2048, 256, 0, stream>>>(Wv,       WTv,    CD, CD, CD);
    wconv_kernel<<<2048, 256, 0, stream>>>(Wg,       WTg,    CD, CD, CD);
    wconv_kernel<<<2048, 256, 0, stream>>>(Wo,       WTo,    CD, CD, CD);
    wconv_kernel<<<2048, 256, 0, stream>>>(gate_w,   WTgate, CD, CD, CD);
    wconv_kernel<<<512,  256, 0, stream>>>(maa_w1,   WTm1,   CD, CL, CLP);
    wconv_kernel<<<256,  256, 0, stream>>>(decay_w1, WTd1,   CD, 64, 64);
    wconv_kernel<<<256,  256, 0, stream>>>(decay_w2, WTd2,   64, CD, CD);

    // 1) conv / dx / xxx
    prep_kernel<<<4096, 256, 0, stream>>>(x, conv_w, maa_x, dxf, dxb, xxxb);

    const dim3 blk(256);
    const dim3 gFull((CD + 63) / 64, CBT / 64);        // N=1024 GEMMs
    const dim3 gLora((CL + 63) / 64, CBT / 64);        // N=160
    const dim3 gU(1, CBT / 64);                        // N=64

    // 2) t5 = tanh(xxx @ maa_w1)
    gemm_bf16_kernel<<<gLora, blk, 0, stream>>>(xxxb, WTm1, t5, nullptr,
                                                CBT, CL, CD, nullptr, ACT_TANH, 0);
    // 3) five mixed inputs
    mix_kernel<<<dim3(CD / 16, CBT / 16, 5), 32, 0, stream>>>(
        t5, maa_w2, x, dxf, maa_w, maa_k, maa_v, maa_r, maa_g,
        Xw, Xk, Xv, Xr, Xg);

    // 4) projections
    gemm_bf16_kernel<<<gFull, blk, 0, stream>>>(Xr, WTr, rb, nullptr,
                                                CBT, CD, CD, nullptr, ACT_NONE, 0);
    gemm_bf16_kernel<<<gFull, blk, 0, stream>>>(Xk, WTk, kb, nullptr,
                                                CBT, CD, CD, nullptr, ACT_NONE, 0);
    gemm_bf16_kernel<<<gFull, blk, 0, stream>>>(Xv, WTv, vb, nullptr,
                                                CBT, CD, CD, nullptr, ACT_NONE, 0);
    gemm_bf16_kernel<<<gFull, blk, 0, stream>>>(Xg, WTg, gb, nullptr,
                                                CBT, CD, CD, nullptr, ACT_SILU, 0);
    // decay MLP: u = tanh(xw @ decay_w1) [bf16]; w_log = -exp(time_decay + u @ decay_w2)
    gemm_bf16_kernel<<<gU, blk, 0, stream>>>(Xw, WTd1, nullptr, ub,
                                             CBT, 64, CD, nullptr, ACT_TANH, 1);
    gemm_bf16_kernel<<<gFull, blk, 0, stream>>>(ub, WTd2, wlb, nullptr,
                                                CBT, CD, 64, time_decay, ACT_NEGEXP, 0);
    // output gate = sigmoid(dx @ gate_w + gate_b)
    gemm_bf16_kernel<<<gFull, blk, 0, stream>>>(dxb, WTgate, gateb, nullptr,
                                                CBT, CD, CD, gate_b, ACT_SIGMOID, 0);

    // 5) bidirectional scans (128 independent 64x64 states)
    rwkv_scan_kernel<<<CB * CH, 64, 0, stream>>>(rb, kb, vb, wlb, yb, 0);
    rwkv_scan_kernel<<<CB * CH, 64, 0, stream>>>(rb, kb, vb, wlb, yb, 1);

    // 6) gate * GroupNorm * g  -> bf16
    gn_gate_kernel<<<(CBT * CH) / 8, 256, 0, stream>>>(yb, gateb, gb, ln_w, ln_b, zb);

    // 7) final projection -> d_out (f32)
    gemm_bf16_kernel<<<gFull, blk, 0, stream>>>(zb, WTo, (float*)d_out, nullptr,
                                                CBT, CD, CD, nullptr, ACT_NONE, 0);
}
